// DRMAttention_38878043963547
// MI455X (gfx1250) — compile-verified, matrix operand order checked by reading, not run on hardware
//
#include <hip/hip_runtime.h>
#include <hip/hip_bf16.h>
#include <math.h>

// ---------------------------------------------------------------------------
// DRM (distance-metric) attention for MI455X / gfx1250, wave32 + WMMA + TDM.
//   B=2, T=1024, D=1024, H=16, dh=64, dm=8
// Stages:
//   1) gemm64 (bf16 WMMA) : Q=x@Wq, K=x@Wk (RoPE fused), V=x@Wv -> (B,H,T,64)
//   2) features           : q_m,k_m,G,Gq,qGq -> f/g feature rows (76 padded)
//   3) attn               : dist^2 = f.g via f32 WMMA (G-tile via TDM),
//                           online softmax, P@V bf16 WMMA
//   4) gemm64 mode 0      : out = O@Wo -> d_out
// ---------------------------------------------------------------------------

typedef __attribute__((ext_vector_type(16))) __bf16 v16bf;
typedef __attribute__((ext_vector_type(8)))  __bf16 v8bf;
typedef __attribute__((ext_vector_type(8)))  float  v8f;
typedef __attribute__((ext_vector_type(2)))  float  v2f;
typedef __attribute__((ext_vector_type(4)))  unsigned int u32x4;
typedef __attribute__((ext_vector_type(8)))  int    i32x8;
typedef __attribute__((ext_vector_type(4)))  int    i32x4;

// Load a 16-element bf16 A/B fragment per CDNA5 layout:
// element i -> K = (i/8)*16 + (lane>>4)*8 + (i%8); caller passes &row[(lane>>4)*8].
__device__ __forceinline__ v16bf load_frag(const __bf16* p) {
    v8bf a = *(const v8bf*)(p);        // K base + 0..7
    v8bf b = *(const v8bf*)(p + 16);   // K base + 16..23
    return __builtin_shufflevector(a, b, 0,1,2,3,4,5,6,7,8,9,10,11,12,13,14,15);
}

// ---------------------------------------------------------------------------
// Tensor Data Mover: async 2-D tile load (fp32 rows) global -> LDS.
// Builds a D# per CDNA5 ISA:
//   group0: count=1 | lds_addr | global_addr(57b) | type=2
//   group1: data_size=4B, tensor_dim0=tile_dim0=stride=row_elems,
//           tensor_dim1=tile_dim1=rows.
// Completion via TENSORcnt (s_wait_tensorcnt). EXEC ignored; wave-level op.
// This toolchain exposes the 6-arg builtin:
//   (u32x4 g0, i32x8 g1, i32x4 g2, i32x4 g3, i32x8 g4, i32 cpol)
// ---------------------------------------------------------------------------
__device__ __forceinline__ void tdm_load_2d_f32(unsigned lds_addr,
                                                const void* gptr,
                                                unsigned row_elems,
                                                unsigned rows) {
    unsigned long long ga = (unsigned long long)(uintptr_t)gptr;
    u32x4 g0;
    g0[0] = 1u;                                               // count=1 (valid)
    g0[1] = lds_addr;                                         // LDS byte address
    g0[2] = (unsigned)ga;                                     // global addr lo
    g0[3] = (unsigned)((ga >> 32) & 0x01FFFFFFu) | (2u << 30);// addr hi | type=2
    i32x8 g1;
    g1[0] = (int)(2u << 16);                                  // data_size=4B
    g1[1] = (int)(row_elems << 16);                           // tensor_dim0 lo
    g1[2] = (int)((row_elems >> 16) | (rows << 16));          // td0 hi | td1 lo
    g1[3] = (int)((rows >> 16) | (row_elems << 16));          // td1 hi | tile_dim0
    g1[4] = (int)rows;                                        // tile_dim1 (dim2=0)
    g1[5] = (int)row_elems;                                   // tensor_dim0_stride lo
    g1[6] = 0;                                                // stride hi | td1_stride lo
    g1[7] = 0;
    i32x4 z4 = {0, 0, 0, 0};                                  // groups 2/3 unused (2-D)
    i32x8 z8 = {0, 0, 0, 0, 0, 0, 0, 0};                      // extra group unused
    __builtin_amdgcn_tensor_load_to_lds(g0, g1, z4, z4, z8, 0);
}

// ---------------------------------------------------------------------------
// Kernel 1/4: C[2048,1024] = X[2048,1024] @ W[1024,1024], bf16 WMMA, f32 acc.
// mode 0: write row-major to Out
// mode 1: RoPE epilogue + scatter to (B,H,T,64)    (Q, K)
// mode 2: scatter to (B,H,T,64)                    (V)
// grid (16 n-tiles == heads, 32 m-tiles), block 128 (4 waves).
// ---------------------------------------------------------------------------
__global__ __launch_bounds__(128) void gemm64(const float* __restrict__ X,
                                              const float* __restrict__ W,
                                              float* __restrict__ Out,
                                              int mode) {
    __shared__ alignas(16) __bf16 Xs[64 * 32];   // [m][k]
    __shared__ alignas(16) __bf16 Wt[64 * 32];   // [n][k] (transposed)

    const int tid  = threadIdx.x;
    const int wave = tid >> 5;
    const int lane = tid & 31;
    const int lo   = lane & 15;
    const int hi   = lane >> 4;
    const int mTile = blockIdx.y * 64;
    const int nTile = blockIdx.x * 64;

    v8f acc[4] = {};

    for (int k0 = 0; k0 < 1024; k0 += 32) {
        for (int e = tid; e < 64 * 32; e += 128) {
            int r = e >> 5, c = e & 31;
            Xs[r * 32 + c] = (__bf16)X[(size_t)(mTile + r) * 1024 + k0 + c];
        }
        for (int e = tid; e < 32 * 64; e += 128) {
            int kk = e >> 6, n = e & 63;
            Wt[n * 32 + kk] = (__bf16)W[(size_t)(k0 + kk) * 1024 + nTile + n];
        }
        // Prefetch next K-slab (global_prefetch_b8) while this slab computes.
        if (k0 + 32 < 1024) {
            __builtin_prefetch(&X[(size_t)(mTile + (tid >> 1)) * 1024 + k0 + 32 +
                                  (tid & 1) * 16], 0, 0);
            __builtin_prefetch(&W[(size_t)(k0 + 32 + (tid >> 2)) * 1024 + nTile +
                                  (tid & 3) * 16], 0, 0);
        }
        __syncthreads();

        v16bf a = load_frag(&Xs[(16 * wave + lo) * 32 + hi * 8]);
#pragma unroll
        for (int j = 0; j < 4; ++j) {
            v16bf b = load_frag(&Wt[(16 * j + lo) * 32 + hi * 8]);
            acc[j] = __builtin_amdgcn_wmma_f32_16x16x32_bf16(
                false, a, false, b, (short)0, acc[j], false, false);
        }
        __syncthreads();
    }

    if (mode == 0) {
#pragma unroll
        for (int j = 0; j < 4; ++j)
#pragma unroll
            for (int r = 0; r < 8; ++r) {
                int row = mTile + 16 * wave + r + 8 * hi;
                Out[(size_t)row * 1024 + nTile + 16 * j + lo] = acc[j][r];
            }
        return;
    }

    const int h = blockIdx.x;  // head index (nTile == h*64)
#pragma unroll
    for (int r = 0; r < 8; ++r) {
        int row = mTile + 16 * wave + r + 8 * hi;
        int b = row >> 10, t = row & 1023;
        float* dst = &Out[(((size_t)b * 16 + h) * 1024 + t) * 64];
        if (mode == 2) {
#pragma unroll
            for (int j = 0; j < 4; ++j) dst[16 * j + lo] = acc[j][r];
        } else {  // RoPE: pair (d, d+32) lives in acc[j] / acc[j+2], same VGPR r
#pragma unroll
            for (int j = 0; j < 2; ++j) {
                int n = 16 * j + lo;  // 0..31
                float inv = __expf(-(float)n * 0.28782313662425575f);  // ln(1e4)/32
                float ang = (float)t * inv;
                float s, c;
                __sincosf(ang, &s, &c);
                float x1 = acc[j][r], x2 = acc[j + 2][r];
                dst[n]      = x1 * c - x2 * s;
                dst[n + 32] = x1 * s + x2 * c;
            }
        }
    }
}

// ---------------------------------------------------------------------------
// Kernel 2: per-token metric features. One thread per (b,h,t).
// f_t = [G(64), -2*Gq(8), qGq, 0,0,0] ; g_s = [km⊗km(64), km(8), 1, 0,0,0]
// ---------------------------------------------------------------------------
__global__ __launch_bounds__(256) void features(const float* __restrict__ Q,
                                                const float* __restrict__ Kh,
                                                const float* __restrict__ Wqm,
                                                const float* __restrict__ Wkm,
                                                const float* __restrict__ Wm,
                                                float* __restrict__ Ff,
                                                float* __restrict__ Gg) {
    __shared__ float sWqm[64 * 8], sWkm[64 * 8], sWm[8 * 64];
    for (int i = threadIdx.x; i < 512; i += 256) {
        sWqm[i] = Wqm[i];
        sWkm[i] = Wkm[i];
        sWm[i]  = Wm[i];
    }
    __syncthreads();

    const int idx = blockIdx.x * 256 + threadIdx.x;  // (b*16+h)*1024 + t
    const float* q = &Q[(size_t)idx * 64];
    const float* k = &Kh[(size_t)idx * 64];

    float qm[8], km[8];
#pragma unroll
    for (int m = 0; m < 8; ++m) {
        float aq = 0.f, ak = 0.f;
        for (int d = 0; d < 64; ++d) {
            aq += q[d] * sWqm[d * 8 + m];
            ak += k[d] * sWkm[d * 8 + m];
        }
        qm[m] = 1.f / (1.f + __expf(-aq));
        km[m] = 1.f / (1.f + __expf(-ak));
    }

    float A[64];
#pragma unroll
    for (int n = 0; n < 64; ++n) {
        float a = 0.f;
        for (int m = 0; m < 8; ++m) a += qm[m] * sWm[m * 64 + n];
        A[n] = a;
    }
    float G[64];
#pragma unroll
    for (int d = 0; d < 8; ++d)
        for (int e = 0; e < 8; ++e) {
            float g = 0.f;
            for (int f = 0; f < 8; ++f) g += A[d * 8 + f] * A[e * 8 + f];
            G[d * 8 + e] = g + ((d == e) ? 0.001f : 0.f);
        }
    float Gq[8], qGq = 0.f;
#pragma unroll
    for (int d = 0; d < 8; ++d) {
        float s = 0.f;
        for (int e = 0; e < 8; ++e) s += G[d * 8 + e] * qm[e];
        Gq[d] = s;
        qGq += s * qm[d];
    }

    float* f = &Ff[(size_t)idx * 76];
    for (int i = 0; i < 64; ++i) f[i] = G[i];
    for (int d = 0; d < 8; ++d) f[64 + d] = -2.f * Gq[d];
    f[72] = qGq;
    f[73] = f[74] = f[75] = 0.f;

    float* g = &Gg[(size_t)idx * 76];
    for (int d = 0; d < 8; ++d)
        for (int e = 0; e < 8; ++e) g[d * 8 + e] = km[d] * km[e];
    for (int d = 0; d < 8; ++d) g[64 + d] = km[d];
    g[72] = 1.f;
    g[73] = g[74] = g[75] = 0.f;
}

// ---------------------------------------------------------------------------
// Kernel 3: flash-style causal attention with metric-distance scores.
// grid (T/64, B*H), block 128 (4 waves: wave w owns query rows 16w..16w+15).
// G-feature tile streamed via Tensor Data Mover (async, TENSORcnt).
// dist^2 tile via v_wmma_f32_16x16x4_f32 (K padded 76), P@V via bf16 WMMA.
// ---------------------------------------------------------------------------
__global__ __launch_bounds__(128) void attn(const float* __restrict__ Ff,
                                            const float* __restrict__ Gg,
                                            const float* __restrict__ V,
                                            const float* __restrict__ tempPtr,
                                            float* __restrict__ O) {
    __shared__ alignas(16) float  Fs[64 * 76];
    __shared__ alignas(16) float  Gs[64 * 76];
    __shared__ alignas(16) __bf16 Vt[64 * 64];  // [d][s] (transposed)
    __shared__ alignas(16) __bf16 Ps[64 * 64];  // [m][s]

    const int bh    = blockIdx.y;
    const int tTile = blockIdx.x * 64;
    const int tid   = threadIdx.x;
    const int wave  = tid >> 5;
    const int lane  = tid & 31;
    const int lo    = lane & 15;
    const int hi    = lane >> 4;
    const float temp = fmaxf(tempPtr[0], 0.1f);
    const unsigned gs_lds = (unsigned)(uintptr_t)(&Gs[0]);

    const float* Fb = &Ff[((size_t)bh * 1024 + tTile) * 76];
    for (int e = tid; e < 64 * 76; e += 128) Fs[e] = Fb[e];
    __syncthreads();

    // Hoist the s-invariant A-side (query feature) fragments into registers:
    // 19 K-steps of 4, element i -> K = 4k + (lane>>4)*2 + i.
    v2f afr[19];
#pragma unroll
    for (int k = 0; k < 19; ++k) {
        const int mrow = (16 * wave + lo) * 76 + 4 * k + hi * 2;
        afr[k][0] = Fs[mrow];
        afr[k][1] = Fs[mrow + 1];
    }

    v8f o[4] = {};
    float rowmax[8], rowsum[8];
#pragma unroll
    for (int r = 0; r < 8; ++r) { rowmax[r] = -INFINITY; rowsum[r] = 0.f; }

    const int nS = blockIdx.x + 1;  // causal: s-tiles 0..tTile
    for (int sb = 0; sb < nS; ++sb) {
        const float* Gb = &Gg[((size_t)bh * 1024 + sb * 64) * 76];
        const float* Vb = &V[((size_t)bh * 1024 + sb * 64) * 64];
        __syncthreads();  // previous iteration's readers are done with Gs/Vt

        // TDM: async DMA of the 64x76 fp32 G-feature tile into LDS (wave 0).
        if (wave == 0) tdm_load_2d_f32(gs_lds, Gb, 76u, 64u);

        // Meanwhile all waves stage V transposed as bf16 for P@V B-fragments.
        for (int e = tid; e < 64 * 64; e += 128) {
            int s = e >> 6, d = e & 63;
            Vt[d * 64 + s] = (__bf16)Vb[s * 64 + d];
        }
        if (wave == 0) __builtin_amdgcn_s_wait_tensorcnt(0);
        __syncthreads();

        // ---- scores: S[m][n] = f_m . g_n  (fp32 WMMA, K steps of 4) ----
        v8f sacc[4] = {};
        for (int k = 0; k < 76; k += 4) {
            v2f a = afr[k >> 2];
#pragma unroll
            for (int j = 0; j < 4; ++j) {
                const int nrow = (16 * j + lo) * 76 + k + hi * 2;
                v2f b;
                b[0] = Gs[nrow];
                b[1] = Gs[nrow + 1];
                sacc[j] = __builtin_amdgcn_wmma_f32_16x16x4_f32(
                    false, a, false, b, (short)0, sacc[j], false, false);
            }
        }

        // ---- mask + scale ----
        const bool diag = (sb == blockIdx.x);
        float p[4][8];
#pragma unroll
        for (int j = 0; j < 4; ++j)
#pragma unroll
            for (int r = 0; r < 8; ++r) {
                float sc = -fmaxf(sacc[j][r], 0.f) / temp;
                if (diag) {
                    int m = 16 * wave + r + 8 * hi;
                    int n = 16 * j + lo;
                    if (n > m) sc = -INFINITY;
                }
                p[j][r] = sc;
            }

        // ---- online softmax (rows striped over 16-lane halves) ----
#pragma unroll
        for (int r = 0; r < 8; ++r) {
            float mx = fmaxf(fmaxf(p[0][r], p[1][r]), fmaxf(p[2][r], p[3][r]));
            for (int off = 8; off >= 1; off >>= 1)
                mx = fmaxf(mx, __shfl_xor(mx, off, 32));
            float nm   = fmaxf(rowmax[r], mx);
            float corr = __expf(rowmax[r] - nm);
            rowmax[r]  = nm;
            rowsum[r] *= corr;
            float sum = 0.f;
#pragma unroll
            for (int j = 0; j < 4; ++j) {
                float e_ = __expf(p[j][r] - nm);
                p[j][r]  = e_;
                sum += e_;
            }
            for (int off = 8; off >= 1; off >>= 1)
                sum += __shfl_xor(sum, off, 32);
            rowsum[r] += sum;
#pragma unroll
            for (int j = 0; j < 4; ++j) o[j][r] *= corr;
        }

        // ---- P tile -> LDS (bf16), wave-private rows ----
#pragma unroll
        for (int j = 0; j < 4; ++j)
#pragma unroll
            for (int r = 0; r < 8; ++r) {
                int m = 16 * wave + r + 8 * hi;
                int n = 16 * j + lo;
                Ps[m * 64 + n] = (__bf16)p[j][r];
            }

        // ---- O += P @ V (bf16 WMMA, K = 64 in two 32 steps) ----
#pragma unroll
        for (int kk = 0; kk < 2; ++kk) {
            v16bf a = load_frag(&Ps[(16 * wave + lo) * 64 + kk * 32 + hi * 8]);
#pragma unroll
            for (int j = 0; j < 4; ++j) {
                v16bf b = load_frag(&Vt[(16 * j + lo) * 64 + kk * 32 + hi * 8]);
                o[j] = __builtin_amdgcn_wmma_f32_16x16x32_bf16(
                    false, a, false, b, (short)0, o[j], false, false);
            }
        }
    }

    // ---- normalize + scatter to (B*T, 1024) for the output projection ----
    const int h = bh & 15, b = bh >> 4;
#pragma unroll
    for (int r = 0; r < 8; ++r) {
        float inv = 1.f / rowsum[r];
        int t = tTile + 16 * wave + r + 8 * hi;
        float* dst = &O[((size_t)(b * 1024 + t)) * 1024 + h * 64];
#pragma unroll
        for (int j = 0; j < 4; ++j) dst[16 * j + lo] = o[j][r] * inv;
    }
}

// ---------------------------------------------------------------------------
extern "C" void kernel_launch(void* const* d_in, const int* in_sizes, int n_in,
                              void* d_out, int out_size, void* d_ws, size_t ws_size,
                              hipStream_t stream) {
    const float* x    = (const float*)d_in[0];
    const float* Wq   = (const float*)d_in[1];
    const float* Wk   = (const float*)d_in[2];
    const float* Wv   = (const float*)d_in[3];
    const float* Wo   = (const float*)d_in[4];
    const float* Wqm  = (const float*)d_in[5];
    const float* Wkm  = (const float*)d_in[6];
    const float* Wm   = (const float*)d_in[7];
    const float* temp = (const float*)d_in[8];

    float* ws = (float*)d_ws;
    const size_t BT = 2048, BHT = 32768;
    float* Q  = ws;                 // (B,H,T,64)
    float* K  = Q + BT * 1024;
    float* V  = K + BT * 1024;
    float* Ff = V + BT * 1024;      // (B*H*T, 76)
    float* Gg = Ff + BHT * 76;
    float* O  = Gg + BHT * 76;      // (B*T, 1024)

    dim3 g(16, 32);
    gemm64<<<g, 128, 0, stream>>>(x, Wq, Q, 1);
    gemm64<<<g, 128, 0, stream>>>(x, Wk, K, 1);
    gemm64<<<g, 128, 0, stream>>>(x, Wv, V, 2);
    features<<<128, 256, 0, stream>>>(Q, K, Wqm, Wkm, Wm, Ff, Gg);
    attn<<<dim3(16, 32), 128, 0, stream>>>(Ff, Gg, V, temp, O);
    gemm64<<<g, 128, 0, stream>>>(O, Wo, (float*)d_out, 0);
}
